// Model_4690104287945
// MI455X (gfx1250) — compile-verified, matrix operand order checked by reading, not run on hardware
//
#include <hip/hip_runtime.h>
#include <hip/hip_bf16.h>
#include <math.h>

typedef float v2f  __attribute__((ext_vector_type(2)));
typedef float v8f  __attribute__((ext_vector_type(8)));
typedef __bf16 v16bf __attribute__((ext_vector_type(16)));

#define WMMA_F32(a,b,c)  __builtin_amdgcn_wmma_f32_16x16x4_f32(false,(a),false,(b),(short)0,(c),false,false)
#define WMMA_BF16(a,b,c) __builtin_amdgcn_wmma_f32_16x16x32_bf16(false,(a),false,(b),(short)0,(c),false,false)

#if __has_builtin(__builtin_amdgcn_global_load_async_to_lds_b128) && \
    __has_builtin(__builtin_amdgcn_s_wait_asynccnt)
#define HAVE_ASYNC_LDS 1
typedef int v4i_t __attribute__((vector_size(16)));
typedef __attribute__((address_space(1))) v4i_t* gptr4_t;
typedef __attribute__((address_space(3))) v4i_t* lptr4_t;
#else
#define HAVE_ASYNC_LDS 0
#endif

// Problem constants
#define BATCH 16
#define SEQ   512
#define CIN   32
#define NORD  256
#define MODES 32
#define BC    512          // BATCH*CIN
#define ROWS  131072       // BC*NORD
#define TCH   64           // t-chunk for cs buffer
#define NCHUNK (SEQ/TCH)   // 8

__device__ __forceinline__ unsigned short f2bf(float x) {
  unsigned int u = __float_as_uint(x);
  unsigned int r = (u + 0x7fffu + ((u >> 16) & 1u)) >> 16;
  return (unsigned short)r;
}

// ---------------- small prep kernels ----------------

__global__ void __launch_bounds__(256) meanstd_kernel(const float* __restrict__ x,
                                                      float* __restrict__ mean,
                                                      float* __restrict__ stdv) {
  int i = blockIdx.x * blockDim.x + threadIdx.x;   // 0..511 = (b,c)
  if (i >= BC) return;
  int b = i >> 5, c = i & 31;
  float s = 0.f, ss = 0.f;
  for (int t = 0; t < SEQ; ++t) {
    float v = x[((size_t)b * SEQ + t) * CIN + c];
    s += v; ss += v * v;
  }
  float m = s * (1.f / SEQ);
  float var = ss * (1.f / SEQ) - m * m;
  mean[i] = m;
  stdv[i] = sqrtf(var + 1e-5f);
}

__global__ void __launch_bounds__(256) norm_kernel(const float* __restrict__ x,
                                                   const float* __restrict__ mean,
                                                   const float* __restrict__ stdv,
                                                   const float* __restrict__ aw,
                                                   const float* __restrict__ ab,
                                                   float* __restrict__ xn) {
  int i = blockIdx.x * 256 + threadIdx.x;          // over B*L*C = 262144
  int c = i & 31;
  int b = i >> 14;                                  // i / (SEQ*CIN)
  int bc = b * 32 + c;
  xn[i] = (x[i] - mean[bc]) / stdv[bc] * aw[c] + ab[c];
}

__global__ void __launch_bounds__(256) initF_kernel(unsigned short* __restrict__ F) {
  int i = blockIdx.x * 256 + threadIdx.x;          // 64*512
  if (i >= 64 * SEQ) return;
  int k = i >> 9, t = i & 511;
  float ang = 6.28318530717958647693f * (float)((k & 31) * t) / (float)SEQ;
  float v = (k < 32) ? cosf(ang) : -sinf(ang);
  F[i] = f2bf(v);
}

__global__ void __launch_bounds__(256) initAt_kernel(const float* __restrict__ A,
                                                     float* __restrict__ At) {
  int i = blockIdx.x * 256 + threadIdx.x;          // 65536
  int m = i >> 8, n = i & 255;
  At[i] = A[n * NORD + m];                          // At[m][n] = A[n][m]
}

// Fold irfft(t=511) phases + 2x mode weights + 1/L into the complex mix weights.
__global__ void __launch_bounds__(256) initG_kernel(const float* __restrict__ wr,
                                                    const float* __restrict__ wi,
                                                    float* __restrict__ G) {
  int i = blockIdx.x * 256 + threadIdx.x;          // 32*256*256
  if (i >= MODES * NORD * NORD) return;
  int k = i >> 16, n = (i >> 8) & 255, o = i & 255;
  float ph = 6.28318530717958647693f * (float)(k * (SEQ - 1)) / (float)SEQ;
  float cph = cosf(ph), sph = sinf(ph);
  float al = ((k == 0) ? 1.f : 2.f) / (float)SEQ;
  float Wr = wr[((size_t)n * NORD + o) * MODES + k];
  float Wi = wi[((size_t)n * NORD + o) * MODES + k];
  G[((size_t)(k * 2 + 0) * NORD + n) * NORD + o] =  al * (Wr * cph - Wi * sph);
  G[((size_t)(k * 2 + 1) * NORD + n) * NORD + o] = -al * (Wi * cph + Wr * sph);
}

__global__ void __launch_bounds__(256) zero_kernel(float* p, int n) {
  int i = blockIdx.x * 256 + threadIdx.x;
  if (i < n) p[i] = 0.f;
}

// ---------------- HiPPO scan: fp32 WMMA, A^T resident in 320KB LDS ----------------
// One WG = 16 rows (one batch, 16 channels). 8 waves, each 2 column tiles.
// S_new[16x256] = S_old[16x256] @ At[256x256] + f_t (x) Bv

__global__ void __launch_bounds__(256) scan_kernel(const float* __restrict__ g_At,
                                                   const float* __restrict__ g_Bv,
                                                   const float* __restrict__ g_xn,
                                                   float* __restrict__ g_state,
                                                   unsigned short* __restrict__ g_cs,
                                                   int t0, int nsteps) {
  extern __shared__ float sm[];
  float* At = sm;                     // [256][260] padded
  float* S  = sm + 256 * 260;        // 2 x [16][257] padded
  const int tid = threadIdx.x;
  const int r0 = blockIdx.x * 16;

#if HAVE_ASYNC_LDS
  // Direct global->LDS async copy of A^T (bypasses VGPRs, tracked by ASYNCcnt).
  // 256 rows; 16 b128 chunks (first 256 B of each 1KB row) copied per row; 4096 chunks over 256 threads.
  for (int i = tid; i < 256 * 16; i += 256) {
    int m = i >> 4, seg = i & 15;
    __builtin_amdgcn_global_load_async_to_lds_b128(
        (gptr4_t)(void*)(g_At + (size_t)m * NORD + seg * 4),
        (lptr4_t)(void*)(At + (size_t)m * 260 + seg * 4),
        0, 0);
  }
#else
  for (int i = tid; i < NORD * NORD; i += 256) {
    int m = i >> 8, n = i & 255;
    At[m * 260 + n] = g_At[i];
  }
#endif
  for (int i = tid; i < 16 * NORD; i += 256) {
    int m = i >> 8, n = i & 255;
    S[m * 257 + n] = g_state[(size_t)(r0 + m) * NORD + n];
  }
#if HAVE_ASYNC_LDS
  __builtin_amdgcn_s_wait_asynccnt(0);
#endif
  __syncthreads();

  const int lane = tid & 31;
  const int wave = tid >> 5;
  const int hi = lane >> 4;
  const int lan = lane & 15;
  const int n0 = wave * 32;
  const float bv0 = g_Bv[n0 + lan];
  const float bv1 = g_Bv[n0 + 16 + lan];
  const int b_idx = r0 >> 5;
  const int c0 = r0 & 31;

  int par = 0;
  for (int ts = 0; ts < nsteps; ++ts) {
    const int t = t0 + ts;
    if (ts + 8 < nsteps)  // prefetch the drive signal a few steps ahead
      __builtin_prefetch(g_xn + ((size_t)b_idx * SEQ + t + 8) * CIN, 0, 1);
    const float* cur = S + par * (16 * 257);
    float* nxt = S + (par ^ 1) * (16 * 257);
    v8f acc0 = {}; v8f acc1 = {};
    #pragma unroll 8
    for (int kk = 0; kk < 64; ++kk) {
      const int k0 = kk * 4;
      v2f a;
      a.x = cur[lan * 257 + k0 + hi * 2];
      a.y = cur[lan * 257 + k0 + hi * 2 + 1];
      v2f b0, b1;
      const float* Arow0 = At + (size_t)(k0 + hi * 2) * 260;
      b0.x = Arow0[n0 + lan];        b1.x = Arow0[n0 + 16 + lan];
      b0.y = Arow0[260 + n0 + lan];  b1.y = Arow0[260 + n0 + 16 + lan];
      acc0 = WMMA_F32(a, b0, acc0);
      acc1 = WMMA_F32(a, b1, acc1);
    }
    const float* fx = g_xn + ((size_t)b_idx * SEQ + t) * CIN + c0 + hi * 8;
    unsigned short* csrow = g_cs + (size_t)ts * ROWS;
    #pragma unroll
    for (int v = 0; v < 8; ++v) {
      const int M = hi * 8 + v;
      float f = fx[v];
      float o0 = acc0[v] + f * bv0;
      float o1 = acc1[v] + f * bv1;
      nxt[M * 257 + n0 + lan] = o0;
      nxt[M * 257 + n0 + 16 + lan] = o1;
      size_t base = (size_t)(r0 + M) * NORD;
      csrow[base + n0 + lan] = f2bf(o0);
      csrow[base + n0 + 16 + lan] = f2bf(o1);
    }
    par ^= 1;
    __syncthreads();
  }
  const float* fin = S + par * (16 * 257);
  for (int i = tid; i < 16 * NORD; i += 256) {
    int m = i >> 8, n = i & 255;
    g_state[(size_t)(r0 + m) * NORD + n] = fin[m * 257 + n];
  }
}

// ---------------- DFT over t (bf16 WMMA): X[64][ROWS] += F[64][t-chunk] @ cs[t-chunk][ROWS]
__global__ void __launch_bounds__(256) dft_kernel(const unsigned short* __restrict__ g_F,
                                                  const unsigned short* __restrict__ g_cs,
                                                  float* __restrict__ g_X,
                                                  int t0, int accumulate) {
  const int tid = threadIdx.x, lane = tid & 31, wave = tid >> 5;
  const int hi = lane >> 4, lan = lane & 15;
  const size_t nbase = (size_t)blockIdx.x * 128 + wave * 16;
  v8f acc[4] = {{0.f}, {0.f}, {0.f}, {0.f}};
  for (int kt = 0; kt < TCH / 32; ++kt) {
    const int tb = kt * 32;
    union { unsigned short u[16]; v16bf v; } Bf;
    #pragma unroll
    for (int p = 0; p < 8; ++p) {
      int kk = ((p < 4) ? p * 2 : 16 + (p - 4) * 2) + hi * 8;
      Bf.u[2 * p]     = g_cs[(size_t)(tb + kk) * ROWS + nbase + lan];
      Bf.u[2 * p + 1] = g_cs[(size_t)(tb + kk + 1) * ROWS + nbase + lan];
    }
    #pragma unroll
    for (int mt = 0; mt < 4; ++mt) {
      union { unsigned short u[16]; v16bf v; } Af;
      const unsigned short* Fr = g_F + (size_t)(mt * 16 + lan) * SEQ + t0 + tb;
      #pragma unroll
      for (int p = 0; p < 8; ++p) {
        int kk = ((p < 4) ? p * 2 : 16 + (p - 4) * 2) + hi * 8;
        Af.u[2 * p] = Fr[kk];
        Af.u[2 * p + 1] = Fr[kk + 1];
      }
      acc[mt] = WMMA_BF16(Af.v, Bf.v, acc[mt]);
    }
  }
  #pragma unroll
  for (int mt = 0; mt < 4; ++mt)
    #pragma unroll
    for (int v = 0; v < 8; ++v) {
      int M = mt * 16 + hi * 8 + v;
      size_t idx = (size_t)M * ROWS + nbase + lan;
      g_X[idx] = accumulate ? (g_X[idx] + acc[mt][v]) : acc[mt][v];
    }
}

// ---------------- mode mix (fp32 WMMA): xdec_part[kg] = sum_k in group (Xr@Gr + Xi@Gi')
__global__ void __launch_bounds__(256) mix_kernel(const float* __restrict__ g_X,
                                                  const float* __restrict__ g_G,
                                                  float* __restrict__ g_part) {
  const int tid = threadIdx.x, lane = tid & 31, wave = tid >> 5;
  const int hi = lane >> 4, lan = lane & 15;
  const int rb = blockIdx.x >> 1, cb = blockIdx.x & 1;  // 8 row-blocks x 2 col-blocks
  const int kg = blockIdx.y;                            // 8 k-groups of 4 modes
  const int mt = wave >> 1;
  const int obase = cb * 128 + (wave & 1) * 64;
  const int mrow = rb * 64 + mt * 16;
  v8f acc[4] = {{0.f}, {0.f}, {0.f}, {0.f}};
  for (int k = kg * 4; k < kg * 4 + 4; ++k) {
    for (int part = 0; part < 2; ++part) {
      const float* X = g_X + (size_t)(k + part * 32) * ROWS;
      const float* G = g_G + (size_t)(k * 2 + part) * (NORD * NORD);
      for (int kk = 0; kk < 64; ++kk) {
        const int k0 = kk * 4;
        v2f a;
        const float* ap = X + (size_t)(mrow + lan) * NORD + k0 + hi * 2;
        a.x = ap[0]; a.y = ap[1];
        #pragma unroll
        for (int q = 0; q < 4; ++q) {
          const float* bp = G + (size_t)(k0 + hi * 2) * NORD + obase + q * 16 + lan;
          v2f b; b.x = bp[0]; b.y = bp[NORD];
          acc[q] = WMMA_F32(a, b, acc[q]);
        }
      }
    }
  }
  #pragma unroll
  for (int q = 0; q < 4; ++q)
    #pragma unroll
    for (int v = 0; v < 8; ++v) {
      int row = mrow + hi * 8 + v;
      int col = obase + q * 16 + lan;
      g_part[((size_t)kg * BC + row) * NORD + col] = acc[q][v];
    }
}

__global__ void __launch_bounds__(256) reduce_kernel(const float* __restrict__ part,
                                                     float* __restrict__ xdec) {
  int i = blockIdx.x * 256 + threadIdx.x;              // BC*NORD = 131072
  float s = 0.f;
  #pragma unroll
  for (int g = 0; g < 8; ++g) s += part[(size_t)g * BC * NORD + i];
  xdec[i] = s;
}

// ---------------- dec = xdec @ E_slice^T (fp32 WMMA) ----------------
__global__ void __launch_bounds__(256) dec_kernel(const float* __restrict__ g_xdec,
                                                  const float* __restrict__ g_E,
                                                  int rowoff,
                                                  float* __restrict__ g_dec) {
  const int tid = threadIdx.x, lane = tid & 31, wave = tid >> 5;
  const int hi = lane >> 4, lan = lane & 15;
  const int rb = blockIdx.x >> 3, cb = blockIdx.x & 7; // 4 x 8 blocks
  const int mrow = rb * 128 + wave * 16;
  const int pbase = cb * 64;
  v8f acc[4] = {{0.f}, {0.f}, {0.f}, {0.f}};
  for (int kk = 0; kk < 64; ++kk) {
    const int k0 = kk * 4;
    v2f a;
    const float* ap = g_xdec + (size_t)(mrow + lan) * NORD + k0 + hi * 2;
    a.x = ap[0]; a.y = ap[1];
    #pragma unroll
    for (int q = 0; q < 4; ++q) {
      const float* bp = g_E + (size_t)(rowoff + pbase + q * 16 + lan) * NORD + k0 + hi * 2;
      v2f b; b.x = bp[0]; b.y = bp[1];
      acc[q] = WMMA_F32(a, b, acc[q]);
    }
  }
  #pragma unroll
  for (int q = 0; q < 4; ++q)
    #pragma unroll
    for (int v = 0; v < 8; ++v)
      g_dec[(size_t)(mrow + hi * 8 + v) * SEQ + pbase + q * 16 + lan] = acc[q][v];
}

// ---------------- final: mlp over scales, denorm, channel projection ----------------
__global__ void __launch_bounds__(32) final_kernel(const float* __restrict__ dec0,
                                                   const float* __restrict__ dec1,
                                                   const float* __restrict__ dec2,
                                                   const float* __restrict__ mlp_w,
                                                   const float* __restrict__ mlp_b,
                                                   const float* __restrict__ aw,
                                                   const float* __restrict__ ab,
                                                   const float* __restrict__ mean,
                                                   const float* __restrict__ stdv,
                                                   const float* __restrict__ proj_w,
                                                   const float* __restrict__ proj_b,
                                                   float* __restrict__ out) {
  const int b = blockIdx.x >> 9, p = blockIdx.x & 511;
  const int c = threadIdx.x;                           // 0..31 (also used as co)
  __shared__ float y[CIN];
  size_t di = (size_t)(b * CIN + c) * SEQ + p;
  float v = dec0[di] * mlp_w[0] + dec1[di] * mlp_w[1] + dec2[di] * mlp_w[2] + mlp_b[0];
  v = (v - ab[c]) / (aw[c] + 1e-10f);
  v = v * stdv[b * CIN + c] + mean[b * CIN + c];
  y[c] = v;
  __syncthreads();
  float s = proj_b[c];
  #pragma unroll
  for (int cc = 0; cc < CIN; ++cc) s += y[cc] * proj_w[cc * CIN + c];
  out[((size_t)b * SEQ + p) * CIN + c] = s;
}

// ---------------- host-side orchestration ----------------

static inline size_t alignup(size_t x) { return (x + 255) & ~(size_t)255; }

extern "C" void kernel_launch(void* const* d_in, const int* in_sizes, int n_in,
                              void* d_out, int out_size, void* d_ws, size_t ws_size,
                              hipStream_t stream) {
  (void)in_sizes; (void)n_in; (void)out_size; (void)ws_size;
  const float* x_enc = (const float*)d_in[0];
  const float* aff_w = (const float*)d_in[4];
  const float* aff_b = (const float*)d_in[5];
  const float* wr[3] = {(const float*)d_in[6], (const float*)d_in[8], (const float*)d_in[10]};
  const float* wi[3] = {(const float*)d_in[7], (const float*)d_in[9], (const float*)d_in[11]};
  const float* mlp_w = (const float*)d_in[12];
  const float* mlp_b = (const float*)d_in[13];
  const float* proj_w = (const float*)d_in[14];
  const float* proj_b = (const float*)d_in[15];
  const float* Amat[3] = {(const float*)d_in[16], (const float*)d_in[19], (const float*)d_in[22]};
  const float* Bv[3]   = {(const float*)d_in[17], (const float*)d_in[20], (const float*)d_in[23]};
  const float* Emat[3] = {(const float*)d_in[18], (const float*)d_in[21], (const float*)d_in[24]};
  const int Eoff[3] = {0, 512, 1536};                  // (512*m - 512) rows

  // workspace carve
  char* w = (char*)d_ws;
  size_t off = 0;
  float* xn   = (float*)(w + off); off = alignup(off + (size_t)BATCH * SEQ * CIN * 4);
  float* mean = (float*)(w + off); off = alignup(off + BC * 4);
  float* stdv = (float*)(w + off); off = alignup(off + BC * 4);
  unsigned short* F = (unsigned short*)(w + off); off = alignup(off + (size_t)64 * SEQ * 2);
  float* At   = (float*)(w + off); off = alignup(off + (size_t)NORD * NORD * 4);
  float* G    = (float*)(w + off); off = alignup(off + (size_t)MODES * 2 * NORD * NORD * 4);
  float* state = (float*)(w + off); off = alignup(off + (size_t)BC * NORD * 4);
  unsigned short* cs = (unsigned short*)(w + off); off = alignup(off + (size_t)TCH * ROWS * 2);
  float* X    = (float*)(w + off); off = alignup(off + (size_t)64 * ROWS * 4);
  float* part = (float*)(w + off); off = alignup(off + (size_t)8 * BC * NORD * 4);
  float* xdec = (float*)(w + off); off = alignup(off + (size_t)BC * NORD * 4);
  float* dec  = (float*)(w + off); off = alignup(off + (size_t)3 * BC * SEQ * 4);

  const size_t scan_smem = (256 * 260 + 2 * 16 * 257) * sizeof(float); // ~292KB < 320KB LDS

  meanstd_kernel<<<2, 256, 0, stream>>>(x_enc, mean, stdv);
  norm_kernel<<<(BATCH * SEQ * CIN) / 256, 256, 0, stream>>>(x_enc, mean, stdv, aff_w, aff_b, xn);
  initF_kernel<<<(64 * SEQ) / 256, 256, 0, stream>>>(F);

  for (int s = 0; s < 3; ++s) {
    initAt_kernel<<<(NORD * NORD) / 256, 256, 0, stream>>>(Amat[s], At);
    initG_kernel<<<(MODES * NORD * NORD) / 256, 256, 0, stream>>>(wr[s], wi[s], G);
    zero_kernel<<<(BC * NORD) / 256, 256, 0, stream>>>(state, BC * NORD);
    for (int ch = 0; ch < NCHUNK; ++ch) {
      scan_kernel<<<BC / 16, 256, scan_smem, stream>>>(At, Bv[s], xn, state, cs, ch * TCH, TCH);
      dft_kernel<<<ROWS / 128, 256, 0, stream>>>(F, cs, X, ch * TCH, ch);
    }
    mix_kernel<<<dim3(16, 8), 256, 0, stream>>>(X, G, part);
    reduce_kernel<<<(BC * NORD) / 256, 256, 0, stream>>>(part, xdec);
    dec_kernel<<<32, 256, 0, stream>>>(xdec, Emat[s], Eoff[s], dec + (size_t)s * BC * SEQ);
  }

  final_kernel<<<BATCH * SEQ, 32, 0, stream>>>(dec, dec + (size_t)BC * SEQ, dec + (size_t)2 * BC * SEQ,
                                               mlp_w, mlp_b, aff_w, aff_b, mean, stdv,
                                               proj_w, proj_b, (float*)d_out);
}
